// MLPPredictor_51711406244225
// MI455X (gfx1250) — matrix operand order repaired and checked by reading, not compiled, not run on hardware
//
#include <hip/hip_runtime.h>

typedef __attribute__((ext_vector_type(16))) __bf16 v16bf;
typedef __attribute__((ext_vector_type(8)))  __bf16 v8bf;
typedef __attribute__((ext_vector_type(4)))  __bf16 v4bf;
typedef __attribute__((ext_vector_type(8)))  float  v8f;
typedef __attribute__((ext_vector_type(4)))  float  v4f;

#define HDIM 128            // hidden size / output features of layer 1
#define KDIM 256            // concat input size (2*H)
#define PITCH 264           // bf16 elements per padded LDS row (256 + 8 pad -> conflict-free)
#define MTILES 2            // 16-row M tiles per wave (32 edges/wave)
#define EDGES_PER_WAVE (16 * MTILES)
#define WAVES 4
#define EDGES_PER_BLOCK (EDGES_PER_WAVE * WAVES)   // 128

__global__ __launch_bounds__(128, 1)
void edge_mlp_wmma(const float* __restrict__ xs, const float* __restrict__ xd,
                   const int*   __restrict__ si, const int*   __restrict__ di,
                   const float* __restrict__ W1, const float* __restrict__ b1,
                   const float* __restrict__ W2, const float* __restrict__ b2,
                   float* __restrict__ out, int n_edges)
{
    // W1 as bf16: 128 rows x 264 (padded) = 67,584 B
    __shared__ __bf16 ldsW[HDIM * PITCH];
    // Per-wave gathered A panels: 4 waves x 32 edges x 264 = 67,584 B
    __shared__ __bf16 ldsA[WAVES * EDGES_PER_WAVE * PITCH];

    const int tid  = threadIdx.x;
    const int lane = tid & 31;
    const int wv   = tid >> 5;
    const int hf   = lane >> 4;    // half-wave: 0 or 1
    const int l16  = lane & 15;

    // ---- Stage W1 (fp32 -> bf16) into LDS, whole block cooperatively ----
    #pragma unroll 4
    for (int i = 0; i < 64; ++i) {
        const int flat = (i * 128 + tid) * 4;
        const int row  = flat >> 8;       // /256
        const int col  = flat & 255;
        v4f w = *(const v4f*)(W1 + flat);
        v4bf wb;
        wb[0] = (__bf16)w[0]; wb[1] = (__bf16)w[1];
        wb[2] = (__bf16)w[2]; wb[3] = (__bf16)w[3];
        *(v4bf*)(&ldsW[row * PITCH + col]) = wb;
    }

    // ---- Gather + convert this wave's 32 edges into its A panel ----
    __bf16* aBase = &ldsA[wv * EDGES_PER_WAVE * PITCH];
    const int ebase = blockIdx.x * EDGES_PER_BLOCK + wv * EDGES_PER_WAVE;
    for (int e = 0; e < EDGES_PER_WAVE; ++e) {
        int edge = ebase + e;
        if (edge >= n_edges) edge = n_edges - 1;        // clamp (E % 128 == 0 anyway)
        const size_t s = (size_t)si[edge] * HDIM;       // uniform across wave -> scalar load
        const size_t d = (size_t)di[edge] * HDIM;
        // 32 lanes x float4 = 128 floats (one full node feature row), L2-resident gather
        v4f fs = *(const v4f*)(xs + s + lane * 4);
        v4f fd = *(const v4f*)(xd + d + lane * 4);
        v4bf bs, bd;
        bs[0]=(__bf16)fs[0]; bs[1]=(__bf16)fs[1]; bs[2]=(__bf16)fs[2]; bs[3]=(__bf16)fs[3];
        bd[0]=(__bf16)fd[0]; bd[1]=(__bf16)fd[1]; bd[2]=(__bf16)fd[2]; bd[3]=(__bf16)fd[3];
        *(v4bf*)(aBase + e * PITCH +       lane * 4) = bs;   // cols 0..127  (x_src)
        *(v4bf*)(aBase + e * PITCH + 128 + lane * 4) = bd;   // cols 128..255 (x_dst)
    }
    __syncthreads();

    // ---- GEMM: [32 edges x 256] x [256 x 128]; B tile reused for both M tiles ----
    v8f acc[MTILES][8] = {};
    #pragma unroll
    for (int kt = 0; kt < 8; ++kt) {
        // A 16x32 bf16 tiles, ISA layout: half0 lanes hold K {0-7,16-23}, half1 {8-15,24-31}
        v16bf a[MTILES];
        #pragma unroll
        for (int m = 0; m < MTILES; ++m) {
            const __bf16* ar = aBase + (m * 16 + l16) * PITCH + kt * 32 + hf * 8;
            v8bf a0 = *(const v8bf*)(ar);         // 16B-aligned ds_load_b128
            v8bf a1 = *(const v8bf*)(ar + 16);
            a[m] = __builtin_shufflevector(a0, a1, 0,1,2,3,4,5,6,7,8,9,10,11,12,13,14,15);
        }
        #pragma unroll
        for (int nt = 0; nt < 8; ++nt) {
            // B 32x16 tile: lane = output column n, half0 holds K {0-7,16-23}, half1 {8-15,24-31}
            const __bf16* br = &ldsW[(nt * 16 + l16) * PITCH + kt * 32 + hf * 8];
            v8bf b0  = *(const v8bf*)(br);
            v8bf b1v = *(const v8bf*)(br + 16);
            v16bf b = __builtin_shufflevector(b0, b1v, 0,1,2,3,4,5,6,7,8,9,10,11,12,13,14,15);
            #pragma unroll
            for (int m = 0; m < MTILES; ++m) {
                acc[m][nt] = __builtin_amdgcn_wmma_f32_16x16x32_bf16(
                    /*neg_a=*/false, a[m], /*neg_b=*/false, b,
                    /*c_mod=*/(short)0, acc[m][nt], /*reuse_a=*/false, /*reuse_b=*/false);
            }
        }
    }

    // ---- Epilogue: bias + relu, dot with W2, reduce over 128 outputs ----
    // C/D layout: VGPR r / lane l -> edge m*16 + r + 8*(l>=16), column n = nt*16 + (l&15)
    const float bias2 = b2[0];
    float bb[8], ww[8];
    #pragma unroll
    for (int nt = 0; nt < 8; ++nt) {
        bb[nt] = b1[nt * 16 + l16];
        ww[nt] = W2[nt * 16 + l16];
    }
    #pragma unroll
    for (int m = 0; m < MTILES; ++m) {
        float part[8] = {};
        #pragma unroll
        for (int nt = 0; nt < 8; ++nt) {
            #pragma unroll
            for (int r = 0; r < 8; ++r) {
                float v = acc[m][nt][r] + bb[nt];
                v = v > 0.0f ? v : 0.0f;
                part[r] = fmaf(v, ww[nt], part[r]);
            }
        }
        #pragma unroll
        for (int r = 0; r < 8; ++r) {
            float p = part[r];
            p += __shfl_xor(p, 1, 32);   // xor of bits 0..3 stays within each 16-lane half
            p += __shfl_xor(p, 2, 32);
            p += __shfl_xor(p, 4, 32);
            p += __shfl_xor(p, 8, 32);
            if (l16 == 0) {
                const int edge = ebase + m * 16 + hf * 8 + r;
                if (edge < n_edges) out[edge] = p + bias2;
            }
        }
    }
}

extern "C" void kernel_launch(void* const* d_in, const int* in_sizes, int n_in,
                              void* d_out, int out_size, void* d_ws, size_t ws_size,
                              hipStream_t stream) {
    const float* xs = (const float*)d_in[0];   // x_src  [N, 128]
    const float* xd = (const float*)d_in[1];   // x_dst  [N, 128]
    const int*   si = (const int*)d_in[2];     // src_idx [E]
    const int*   di = (const int*)d_in[3];     // dst_idx [E]
    const float* W1 = (const float*)d_in[4];   // [128, 256]
    const float* b1 = (const float*)d_in[5];   // [128]
    const float* W2 = (const float*)d_in[6];   // [1, 128]
    const float* b2 = (const float*)d_in[7];   // [1]
    float* out = (float*)d_out;                // [E]

    const int n_edges = in_sizes[2];
    const int blocks = (n_edges + EDGES_PER_BLOCK - 1) / EDGES_PER_BLOCK;
    edge_mlp_wmma<<<blocks, 128, 0, stream>>>(xs, xd, si, di, W1, b1, W2, b2, out, n_edges);
}
